// Net_37374805410016
// MI455X (gfx1250) — matrix operand order repaired and checked by reading, not compiled, not run on hardware
//
#include <hip/hip_runtime.h>
#include <hip/hip_bf16.h>
#include <float.h>

// ---------------------------------------------------------------------------
// CDNA5 (gfx1250, wave32) implementation of the SplineCNN pipeline.
// Dense einsums -> v_wmma_f32_16x16x32_bf16; 4 N-tiles per wave with the
// A-operand held in registers across the 4 WMMAs.
// ---------------------------------------------------------------------------

typedef __attribute__((ext_vector_type(16))) __bf16 v16bf;
typedef __attribute__((ext_vector_type(8)))  float  v8f;

union V16Cast { uint4 q[2]; v16bf v; };

__device__ __forceinline__ __bf16 f2bf(float x) {
  unsigned u = __float_as_uint(x);
  unsigned r = (u + 0x7FFFu + ((u >> 16) & 1u)) >> 16;   // round-to-nearest-even
  unsigned short s = (unsigned short)r;
  return __builtin_bit_cast(__bf16, s);
}

__device__ __forceinline__ void atomicMaxFloat(float* addr, float val) {
  // Sign-split trick: monotone int/uint mappings -> native global atomics.
  if (val >= 0.0f) atomicMax((int*)addr, __float_as_int(val));
  else             atomicMin((unsigned int*)addr, __float_as_uint(val));
}

// ------------------------------- fills -------------------------------------

__global__ void fill_f32(float* __restrict__ p, float v, size_t n) {
  size_t i = (size_t)blockIdx.x * blockDim.x + threadIdx.x;
  if (i < n) p[i] = v;
}

__global__ void fill_i32(int* __restrict__ p, int v, size_t n) {
  size_t i = (size_t)blockIdx.x * blockDim.x + threadIdx.x;
  if (i < n) p[i] = v;
}

// --------------------------- bf16 staging ----------------------------------

// x [N][F] f32 -> xh [N][32] bf16, zero-padded columns F..31
__global__ void cvt_x_pad(const float* __restrict__ x, __bf16* __restrict__ xh,
                          int N, int F) {
  int i = blockIdx.x * blockDim.x + threadIdx.x;
  if (i >= N * 32) return;
  int n = i >> 5, k = i & 31;
  float v = (k < F) ? x[n * F + k] : 0.0f;
  xh[i] = f2bf(v);
}

// W [NK][FIN][FOUT], root [FIN][FOUT] -> Bh [(NK*FOUT + FOUT)][32] bf16
// Bh[n][k]: k-contiguous so each lane's 16 B-values are one 32-byte load pair.
__global__ void cvt_w(const float* __restrict__ W, const float* __restrict__ root,
                      __bf16* __restrict__ Bh, int NKF, int FIN, int FOUT) {
  int i = blockIdx.x * blockDim.x + threadIdx.x;
  int total = (NKF + FOUT) * 32;
  if (i >= total) return;
  int n = i >> 5, k = i & 31;
  float v = 0.0f;
  if (k < FIN) {
    if (n < NKF) {
      int ks = n / FOUT, o = n % FOUT;
      v = W[((size_t)ks * FIN + k) * FOUT + o];
    } else {
      int o = n - NKF;
      v = root[k * FOUT + o];
    }
  }
  Bh[i] = f2bf(v);
}

// ------------------------------ WMMA GEMM ----------------------------------
// C[M][ldc] = A[M][32](bf16) * B[ldc][32]^T(bf16).
// One wave handles one 16-row M tile x four consecutive 16-col N tiles:
// A is loaded once (32 B/lane) and reused across 4 v_wmma_f32_16x16x32_bf16.
// ldc must be a multiple of 64 (holds: 4032 = 63*64, 8064 = 126*64).
__global__ void wmma_gemm(const __bf16* __restrict__ A,
                          const __bf16* __restrict__ B,
                          float* __restrict__ C,
                          int Mtiles, int Ngroups, int ldc) {
  int lane = threadIdx.x & 31;
  int wave = threadIdx.x >> 5;
  int grp  = blockIdx.x * (blockDim.x >> 5) + wave;
  if (grp >= Mtiles * Ngroups) return;      // wave-uniform: EXEC stays all-1s
  int m0 = (grp / Ngroups) << 4;
  int n0 = (grp % Ngroups) << 6;            // 4 tiles * 16 columns
  int half = lane >> 4;                     // K-half select per ISA layout
  int lr   = lane & 15;

  // A 16x32 bf16: lane row = m0+lr; elems [0..7]=K(kb..kb+7), [8..15]=K(16+kb..)
  const __bf16* arow = A + (size_t)(m0 + lr) * 32;
  V16Cast ua;
  ua.q[0] = *(const uint4*)(arow + half * 8);
  ua.q[1] = *(const uint4*)(arow + 16 + half * 8);

#pragma unroll
  for (int t = 0; t < 4; ++t) {
    // B 32x16 bf16: lane col = n0+16t+lr; elems [0..7]=K(kb..kb+7), [8..15]=K(16+kb..)
    const __bf16* bcol = B + (size_t)(n0 + t * 16 + lr) * 32 + half * 8;
    V16Cast ub;
    ub.q[0] = *(const uint4*)(bcol);
    ub.q[1] = *(const uint4*)(bcol + 16);

    v8f c = {};
    c = __builtin_amdgcn_wmma_f32_16x16x32_bf16(false, ua.v, false, ub.v,
                                                (short)0, c, false, false);

    // D 16x16 f32: lane col = n0+16t+lr, VGPR r -> row m0 + r + 8*half
    float* crow = C + (size_t)(m0 + half * 8) * ldc + (n0 + t * 16 + lr);
#pragma unroll
    for (int r = 0; r < 8; ++r) crow[(size_t)r * ldc] = c[r];
  }
}

// --------------------- spline message passing (gather) ---------------------
// One thread per (edge, out-channel). Trilinear basis (K=5, DIM=3, 8 corners).
template <int FOUT>
__global__ void edge_msg(const float* __restrict__ xw, int ldn,
                         const int* __restrict__ ei, const float* __restrict__ attr,
                         float* __restrict__ agg, float* __restrict__ deg, int E) {
  int t = blockIdx.x * blockDim.x + threadIdx.x;
  int e = t / FOUT;
  int o = t % FOUT;
  if (e >= E) return;
  int src = ei[e];
  int dst = ei[E + e];

  float v0 = attr[3 * e + 0] * 4.0f; float g0 = floorf(v0); int l0 = (int)g0; float f0 = v0 - g0;
  float v1 = attr[3 * e + 1] * 4.0f; float g1 = floorf(v1); int l1 = (int)g1; float f1 = v1 - g1;
  float v2 = attr[3 * e + 2] * 4.0f; float g2 = floorf(v2); int l2 = (int)g2; float f2 = v2 - g2;

  const float* row = xw + (size_t)src * ldn;
  __builtin_prefetch(row, 0, 0);            // global_prefetch_b8

  float m = 0.0f;
#pragma unroll
  for (int s = 0; s < 8; ++s) {
    int b0 = s & 1, b1 = (s >> 1) & 1, b2 = (s >> 2) & 1;
    float w = (b0 ? f0 : 1.0f - f0) * (b1 ? f1 : 1.0f - f1) * (b2 ? f2 : 1.0f - f2);
    int idx = (l0 + b0) + (l1 + b1) * 5 + (l2 + b2) * 25;
    idx = idx < 124 ? idx : 124;            // clamp (w==0 on the clamped corners)
    m += w * row[idx * FOUT + o];
  }
  atomicAdd(&agg[(size_t)dst * FOUT + o], m);
  if (o == 0) atomicAdd(&deg[dst], 1.0f);
}

// agg/deg normalize + root term (folded into XW extra columns) + bias + ELU
__global__ void finalize_node(const float* __restrict__ agg, const float* __restrict__ deg,
                              const float* __restrict__ xw, int ldn, int rootcol,
                              const float* __restrict__ bias,
                              float* __restrict__ h, int N, int F) {
  int i = blockIdx.x * blockDim.x + threadIdx.x;
  if (i >= N * F) return;
  int n = i / F, o = i % F;
  float d = deg[n]; d = d < 1.0f ? 1.0f : d;
  float a = agg[i] / d + xw[(size_t)n * ldn + rootcol + o] + bias[o];
  h[i] = a > 0.0f ? a : expm1f(a);
}

__global__ void pool_max(const float* __restrict__ h, const int* __restrict__ cluster,
                         float* __restrict__ out, int N, int F) {
  int i = blockIdx.x * blockDim.x + threadIdx.x;
  if (i >= N * F) return;
  int n = i / F, o = i % F;
  atomicMaxFloat(&out[(size_t)cluster[n] * F + o], h[i]);
}

// segment_max fill (-inf in JAX, -FLT_MAX here) + !isfinite -> 0
__global__ void fixup_neg(float* __restrict__ p, size_t n) {
  size_t i = (size_t)blockIdx.x * blockDim.x + threadIdx.x;
  if (i >= n) return;
  float v = p[i];
  if (!(v > -FLT_MAX)) p[i] = 0.0f;
}

__global__ void batch_max(const int* __restrict__ batch2, const int* __restrict__ cluster2,
                          int* __restrict__ batch3, int N) {
  int i = blockIdx.x * blockDim.x + threadIdx.x;
  if (i >= N) return;
  atomicMax(&batch3[cluster2[i]], batch2[i]);   // init 0 == maximum(segmax, 0)
}

__global__ void global_pool(const float* __restrict__ x3, const int* __restrict__ batch3,
                            float* __restrict__ sums, float* __restrict__ cnt, int N3) {
  int i = blockIdx.x * blockDim.x + threadIdx.x;
  if (i >= N3 * 64) return;
  int n = i >> 6, o = i & 63;
  int g = batch3[n];
  atomicAdd(&sums[g * 64 + o], x3[i]);
  if (o == 0) atomicAdd(&cnt[g], 1.0f);
}

// xg = sums/cnt ; h = elu(xg@fc1+b) ; y = h@fc2+b ; log_softmax over 1 class == 0
__global__ void mlp_kernel(const float* __restrict__ sums, const float* __restrict__ cnt,
                           const float* __restrict__ fc1w, const float* __restrict__ fc1b,
                           const float* __restrict__ fc2w, const float* __restrict__ fc2b,
                           float* __restrict__ out) {
  __shared__ float xg[32 * 64];
  __shared__ float h[32 * 128];
  int t = threadIdx.x;
  for (int i = t; i < 32 * 64; i += 128) {
    int g = i >> 6;
    float c = cnt[g]; c = c < 1.0f ? 1.0f : c;
    xg[i] = sums[i] / c;
  }
  __syncthreads();
  for (int idx = t; idx < 32 * 128; idx += 128) {
    int g = idx >> 7, j = idx & 127;
    float acc = fc1b[j];
    for (int i = 0; i < 64; ++i) acc += xg[g * 64 + i] * fc1w[i * 128 + j];
    h[idx] = acc > 0.0f ? acc : expm1f(acc);
  }
  __syncthreads();
  if (t < 32) {
    float y = fc2b[0];
    for (int j = 0; j < 128; ++j) y += h[t * 128 + j] * fc2w[j];
    out[t] = y - y;   // log_softmax(axis=1) of a (G,1) tensor is exactly 0
  }
}

// ---------------------------------------------------------------------------

extern "C" void kernel_launch(void* const* d_in, const int* in_sizes, int n_in,
                              void* d_out, int out_size, void* d_ws, size_t ws_size,
                              hipStream_t stream) {
  const float* x     = (const float*)d_in[0];
  const int*   ei1   = (const int*)  d_in[1];
  const float* attr1 = (const float*)d_in[2];
  const int*   clu1  = (const int*)  d_in[3];
  const int*   ei2   = (const int*)  d_in[4];
  const float* attr2 = (const float*)d_in[5];
  const int*   clu2  = (const int*)  d_in[6];
  const int*   bat2  = (const int*)  d_in[7];
  const float* W1    = (const float*)d_in[8];
  const float* root1 = (const float*)d_in[9];
  const float* b1    = (const float*)d_in[10];
  const float* W2    = (const float*)d_in[11];
  const float* root2 = (const float*)d_in[12];
  const float* b2    = (const float*)d_in[13];
  const float* fc1w  = (const float*)d_in[14];
  const float* fc1b  = (const float*)d_in[15];
  const float* fc2w  = (const float*)d_in[16];
  const float* fc2b  = (const float*)d_in[17];
  float* out = (float*)d_out;

  constexpr int N1 = 30000, N2 = 15000, N3 = 8000, G = 32;
  constexpr int E1 = 480000, E2 = 240000;
  constexpr int N2P = 15008;               // N2 padded to multiple of 16
  constexpr int LD1 = 125 * 32 + 32;       // 4032 (spline cols + root cols)
  constexpr int LD2 = 125 * 64 + 64;       // 8064

  // bump allocator over d_ws (peak ~508 MB; XW reused across layers)
  char* wsp = (char*)d_ws;
  size_t off = 0;
  auto alloc = [&](size_t bytes) -> void* {
    void* p = wsp + off;
    off = (off + bytes + 255) & ~(size_t)255;
    return p;
  };
  size_t xwElems = (size_t)N1 * LD1 > (size_t)N2P * LD2 ? (size_t)N1 * LD1
                                                        : (size_t)N2P * LD2;
  float*  XW   = (float*) alloc(xwElems * 4);
  __bf16* xh1  = (__bf16*)alloc((size_t)N1  * 32 * 2);
  __bf16* Wh1  = (__bf16*)alloc((size_t)LD1 * 32 * 2);
  float*  agg1 = (float*) alloc((size_t)N1 * 32 * 4);
  float*  deg1 = (float*) alloc((size_t)N1 * 4);
  float*  h1   = (float*) alloc((size_t)N1 * 32 * 4);
  float*  x2   = (float*) alloc((size_t)N2P * 32 * 4);
  __bf16* xh2  = (__bf16*)alloc((size_t)N2P * 32 * 2);
  __bf16* Wh2  = (__bf16*)alloc((size_t)LD2 * 32 * 2);
  float*  agg2 = (float*) alloc((size_t)N2P * 64 * 4);
  float*  deg2 = (float*) alloc((size_t)N2P * 4);
  float*  h2   = (float*) alloc((size_t)N2P * 64 * 4);
  float*  x3   = (float*) alloc((size_t)N3 * 64 * 4);
  int*    bat3 = (int*)   alloc((size_t)N3 * 4);
  float*  sums = (float*) alloc((size_t)G * 64 * 4);
  float*  cnt  = (float*) alloc((size_t)G * 4);

  const int T = 256;                        // 8 wave32 waves per block
  auto blks = [&](size_t n) { return dim3((unsigned)((n + T - 1) / T)); };

  // ---------------- Layer 1 ----------------
  cvt_x_pad<<<blks((size_t)N1 * 32), T, 0, stream>>>(x, xh1, N1, 20);
  cvt_w   <<<blks((size_t)LD1 * 32), T, 0, stream>>>(W1, root1, Wh1, 125 * 32, 20, 32);
  {
    int groups = (N1 / 16) * (LD1 / 64);    // 1875 * 63
    wmma_gemm<<<dim3((groups + 7) / 8), T, 0, stream>>>(xh1, Wh1, XW,
                                                        N1 / 16, LD1 / 64, LD1);
  }
  fill_f32<<<blks((size_t)N1 * 32), T, 0, stream>>>(agg1, 0.0f, (size_t)N1 * 32);
  fill_f32<<<blks((size_t)N1), T, 0, stream>>>(deg1, 0.0f, (size_t)N1);
  edge_msg<32><<<blks((size_t)E1 * 32), T, 0, stream>>>(XW, LD1, ei1, attr1,
                                                        agg1, deg1, E1);
  finalize_node<<<blks((size_t)N1 * 32), T, 0, stream>>>(agg1, deg1, XW, LD1,
                                                         125 * 32, b1, h1, N1, 32);
  fill_f32<<<blks((size_t)N2P * 32), T, 0, stream>>>(x2, -FLT_MAX, (size_t)N2P * 32);
  pool_max<<<blks((size_t)N1 * 32), T, 0, stream>>>(h1, clu1, x2, N1, 32);
  fixup_neg<<<blks((size_t)N2P * 32), T, 0, stream>>>(x2, (size_t)N2P * 32);

  // ---------------- Layer 2 ----------------
  cvt_x_pad<<<blks((size_t)N2P * 32), T, 0, stream>>>(x2, xh2, N2P, 32);
  cvt_w   <<<blks((size_t)LD2 * 32), T, 0, stream>>>(W2, root2, Wh2, 125 * 64, 32, 64);
  {
    int groups = (N2P / 16) * (LD2 / 64);   // 938 * 126
    wmma_gemm<<<dim3((groups + 7) / 8), T, 0, stream>>>(xh2, Wh2, XW,
                                                        N2P / 16, LD2 / 64, LD2);
  }
  fill_f32<<<blks((size_t)N2P * 64), T, 0, stream>>>(agg2, 0.0f, (size_t)N2P * 64);
  fill_f32<<<blks((size_t)N2P), T, 0, stream>>>(deg2, 0.0f, (size_t)N2P);
  edge_msg<64><<<blks((size_t)E2 * 64), T, 0, stream>>>(XW, LD2, ei2, attr2,
                                                        agg2, deg2, E2);
  finalize_node<<<blks((size_t)N2 * 64), T, 0, stream>>>(agg2, deg2, XW, LD2,
                                                         125 * 64, b2, h2, N2, 64);
  fill_f32<<<blks((size_t)N3 * 64), T, 0, stream>>>(x3, -FLT_MAX, (size_t)N3 * 64);
  pool_max<<<blks((size_t)N2 * 64), T, 0, stream>>>(h2, clu2, x3, N2, 64);
  fixup_neg<<<blks((size_t)N3 * 64), T, 0, stream>>>(x3, (size_t)N3 * 64);

  // ---------------- Readout ----------------
  fill_i32<<<blks((size_t)N3), T, 0, stream>>>(bat3, 0, (size_t)N3);
  batch_max<<<blks((size_t)N2), T, 0, stream>>>(bat2, clu2, bat3, N2);
  fill_f32<<<blks((size_t)G * 64), T, 0, stream>>>(sums, 0.0f, (size_t)G * 64);
  fill_f32<<<blks((size_t)G), T, 0, stream>>>(cnt, 0.0f, (size_t)G);
  global_pool<<<blks((size_t)N3 * 64), T, 0, stream>>>(x3, bat3, sums, cnt, N3);
  mlp_kernel<<<1, 128, 0, stream>>>(sums, cnt, fc1w, fc1b, fc2w, fc2b, out);

  (void)in_sizes; (void)n_in; (void)out_size; (void)ws_size;
}
